// SwinTransformerBlock_4449586118909
// MI455X (gfx1250) — compile-verified, matrix operand order checked by reading, not run on hardware
//
#include <hip/hip_runtime.h>
#include <cstdint>

// ---------------------------------------------------------------------------
// Problem constants (Swin block: B=16, H=W=56, C=512, NH=16, WS=7)
// ---------------------------------------------------------------------------
constexpr int Hh   = 56;
constexpr int Ww   = 56;
constexpr int Cc   = 512;
constexpr int NHh  = 16;
constexpr int SSs  = 3;
constexpr int Nn   = 49;     // tokens per window
constexpr int NP   = 64;     // padded tokens per window
constexpr int HD   = 32;     // head dim
constexpr int MLPH = 2048;
constexpr int TWIN = 1024;   // total windows (B * 64)
constexpr int MPAD = TWIN * NP;      // 65536 padded rows
constexpr int TOK  = 16 * Hh * Ww;   // 50176 tokens
constexpr float SCALE = 0.17677669529663689f;  // 32^-0.5

typedef __attribute__((ext_vector_type(16))) __bf16 v16bf;
typedef __attribute__((ext_vector_type(8)))  float  v8f;

// ---------------------------------------------------------------------------
// WMMA helpers.  CDNA5 bf16 fragment layout (ISA 7.12.2):
//   row   = lane & 15            (M for A, N for B)
//   khalf = (lane >> 4) * 8
//   elem e -> K = (e & 7) + (e >= 8 ? 16 : 0) + khalf
// => per lane two contiguous 8x bf16 (16B) chunks at +0 and +16.
// ---------------------------------------------------------------------------
__device__ __forceinline__ v16bf frag_ld(const __bf16* p) {
  v16bf f;
#pragma unroll
  for (int e = 0; e < 8; ++e) f[e] = p[e];
#pragma unroll
  for (int e = 0; e < 8; ++e) f[e + 8] = p[e + 16];
  return f;
}

__device__ __forceinline__ v8f wmma_bf16(v16bf a, v16bf b, v8f c) {
  return __builtin_amdgcn_wmma_f32_16x16x32_bf16(false, a, false, b,
                                                 (short)0, c, false, false);
}

// CDNA5 async memory->LDS copy (GLOBAL_LOAD_ASYNC_TO_LDS_B128, ASYNCcnt).
// LDS address operand = low 32 bits of the flat shared-memory address.
__device__ __forceinline__ void async_ld_b128(uint32_t dsoff, const void* g) {
  asm volatile("global_load_async_to_lds_b128 %0, %1, off"
               :
               : "v"(dsoff), "v"(g)
               : "memory");
}
__device__ __forceinline__ void wait_async0() {
  asm volatile("s_wait_asynccnt 0" ::: "memory");
}
template <typename T>
__device__ __forceinline__ uint32_t lds_off(T* p) {
  return (uint32_t)(uintptr_t)(void*)p;
}

// 32x64 output tile per wave: 2 A-frags x 4 B-frags -> 8 WMMAs per K=32 step.
// A: [M][K] bf16 row-major activations.  W: [N][K] bf16 row-major weights
// (matches x @ W.T).
__device__ __forceinline__ void gemm32x64(const __bf16* __restrict__ A,
                                          const __bf16* __restrict__ W,
                                          int K, int m0, int n0, int lane,
                                          v8f acc[2][4]) {
  const int l15 = lane & 15;
  const int kh  = (lane >> 4) << 3;
  const __bf16* a0 = A + (size_t)(m0 + l15) * K + kh;
  const __bf16* a1 = A + (size_t)(m0 + 16 + l15) * K + kh;
  const __bf16* b0 = W + (size_t)(n0 + l15) * K + kh;
  const __bf16* b1 = b0 + (size_t)16 * K;
  const __bf16* b2 = b0 + (size_t)32 * K;
  const __bf16* b3 = b0 + (size_t)48 * K;
  for (int k = 0; k < K; k += 32) {
    __builtin_prefetch(a0 + k + 256, 0, 1);
    __builtin_prefetch(b0 + k + 256, 0, 1);
    v16bf fa0 = frag_ld(a0 + k);
    v16bf fa1 = frag_ld(a1 + k);
    v16bf fb0 = frag_ld(b0 + k);
    v16bf fb1 = frag_ld(b1 + k);
    v16bf fb2 = frag_ld(b2 + k);
    v16bf fb3 = frag_ld(b3 + k);
    acc[0][0] = wmma_bf16(fa0, fb0, acc[0][0]);
    acc[1][0] = wmma_bf16(fa1, fb0, acc[1][0]);
    acc[0][1] = wmma_bf16(fa0, fb1, acc[0][1]);
    acc[1][1] = wmma_bf16(fa1, fb1, acc[1][1]);
    acc[0][2] = wmma_bf16(fa0, fb2, acc[0][2]);
    acc[1][2] = wmma_bf16(fa1, fb2, acc[1][2]);
    acc[0][3] = wmma_bf16(fa0, fb3, acc[0][3]);
    acc[1][3] = wmma_bf16(fa1, fb3, acc[1][3]);
  }
}

// ---------------------------------------------------------------------------
// K0: f32 -> bf16 weight conversion
// ---------------------------------------------------------------------------
__global__ void f32_to_bf16_kernel(const float* __restrict__ s,
                                   __bf16* __restrict__ d, int n) {
  int i = blockIdx.x * 256 + threadIdx.x;
  if (i < n) d[i] = (__bf16)s[i];
}

// ---------------------------------------------------------------------------
// K1: LayerNorm + roll(-3,-3) + window partition -> a1w bf16 [1024][64][512]
// One block per destination (window,row); pad rows 49..63 zeroed so the QKV
// GEMM over padded M is well defined.
// ---------------------------------------------------------------------------
__global__ void ln_roll_part_kernel(const float* __restrict__ x,
                                    const float* __restrict__ g,
                                    const float* __restrict__ b,
                                    __bf16* __restrict__ outp) {
  __shared__ float red[256];
  const int blk = blockIdx.x;
  const int win = blk >> 6, row = blk & 63;
  __bf16* dst = outp + (size_t)blk * Cc;
  if (row >= Nn) {
    for (int i = threadIdx.x; i < Cc; i += 256) dst[i] = (__bf16)0.0f;
    return;
  }
  const int bb = win >> 6, wl = win & 63;
  const int wh = wl >> 3, wwi = wl & 7;
  const int r = row / 7, c = row % 7;
  const int h = (wh * 7 + r + SSs) % Hh;   // inverse of roll(-SS)
  const int w = (wwi * 7 + c + SSs) % Ww;
  const float* src = x + ((size_t)bb * (Hh * Ww) + h * Ww + w) * Cc;
  const int tid = threadIdx.x;
  float s = 0.f;
  for (int i = tid; i < Cc; i += 256) s += src[i];
  red[tid] = s; __syncthreads();
  for (int st = 128; st > 0; st >>= 1) { if (tid < st) red[tid] += red[tid + st]; __syncthreads(); }
  const float mu = red[0] * (1.0f / Cc);
  __syncthreads();
  float vs = 0.f;
  for (int i = tid; i < Cc; i += 256) { float d = src[i] - mu; vs += d * d; }
  red[tid] = vs; __syncthreads();
  for (int st = 128; st > 0; st >>= 1) { if (tid < st) red[tid] += red[tid + st]; __syncthreads(); }
  const float inv = rsqrtf(red[0] * (1.0f / Cc) + 1e-5f);
  for (int i = tid; i < Cc; i += 256)
    dst[i] = (__bf16)((src[i] - mu) * inv * g[i] + b[i]);
}

// ---------------------------------------------------------------------------
// K5: plain LayerNorm (token-major) -> bf16
// ---------------------------------------------------------------------------
__global__ void ln_kernel(const float* __restrict__ x,
                          const float* __restrict__ g,
                          const float* __restrict__ b,
                          __bf16* __restrict__ outp) {
  __shared__ float red[256];
  const float* src = x + (size_t)blockIdx.x * Cc;
  __bf16* dst = outp + (size_t)blockIdx.x * Cc;
  const int tid = threadIdx.x;
  float s = 0.f;
  for (int i = tid; i < Cc; i += 256) s += src[i];
  red[tid] = s; __syncthreads();
  for (int st = 128; st > 0; st >>= 1) { if (tid < st) red[tid] += red[tid + st]; __syncthreads(); }
  const float mu = red[0] * (1.0f / Cc);
  __syncthreads();
  float vs = 0.f;
  for (int i = tid; i < Cc; i += 256) { float d = src[i] - mu; vs += d * d; }
  red[tid] = vs; __syncthreads();
  for (int st = 128; st > 0; st >>= 1) { if (tid < st) red[tid] += red[tid + st]; __syncthreads(); }
  const float inv = rsqrtf(red[0] * (1.0f / Cc) + 1e-5f);
  for (int i = tid; i < Cc; i += 256)
    dst[i] = (__bf16)((src[i] - mu) * inv * g[i] + b[i]);
}

// ---------------------------------------------------------------------------
// K2: QKV GEMM  [65536 x 512] @ [1536 x 512]^T, bias, q pre-scaled by 1/sqrt(d)
// Output routed into per-(window,head) [64][32] tiles for the attention stage.
// ---------------------------------------------------------------------------
__global__ void __launch_bounds__(128)
qkv_gemm_kernel(const __bf16* __restrict__ A, const __bf16* __restrict__ W,
                const float* __restrict__ bias,
                __bf16* __restrict__ q, __bf16* __restrict__ k,
                __bf16* __restrict__ v) {
  const int lane = threadIdx.x & 31, wave = threadIdx.x >> 5;
  const int m0 = blockIdx.x * 32;
  const int n0 = (blockIdx.y * 4 + wave) * 64;
  v8f acc[2][4] = {};
  gemm32x64(A, W, Cc, m0, n0, lane, acc);
  const int l15 = lane & 15, mh = (lane >> 4) << 3;
#pragma unroll
  for (int mi = 0; mi < 2; ++mi)
#pragma unroll
    for (int ni = 0; ni < 4; ++ni)
#pragma unroll
      for (int r = 0; r < 8; ++r) {
        const int gm = m0 + mi * 16 + mh + r;
        const int gn = n0 + ni * 16 + l15;
        float val = acc[mi][ni][r] + bias[gn];
        __bf16* dst; int nn;
        if (gn < Cc)            { dst = q; nn = gn;          val *= SCALE; }
        else if (gn < 2 * Cc)   { dst = k; nn = gn - Cc; }
        else                    { dst = v; nn = gn - 2 * Cc; }
        const int head = nn >> 5, d = nn & 31;
        const int win = gm >> 6, row = gm & 63;
        dst[(((size_t)win * NHh + head) * NP + row) * HD + d] = (__bf16)val;
      }
}

// ---------------------------------------------------------------------------
// K3: per-(window,head) attention.  q/k/v tiles staged to LDS with CDNA5
// GLOBAL_LOAD_ASYNC_TO_LDS_B128.  S = q·k^T (one WMMA per 16x16 tile, K=32),
// + rel-pos bias + shift mask, softmax over 49 keys, O = P @ v (K=64, 2 WMMAs).
// ---------------------------------------------------------------------------
__global__ void __launch_bounds__(128)
attn_kernel(const __bf16* __restrict__ qg, const __bf16* __restrict__ kg,
            const __bf16* __restrict__ vg, const float* __restrict__ rel_tab,
            __bf16* __restrict__ xw) {
  const int win = blockIdx.x, head = blockIdx.y;
  __shared__ __attribute__((aligned(16))) __bf16 qs[NP * HD];
  __shared__ __attribute__((aligned(16))) __bf16 ks[NP * HD];
  __shared__ __attribute__((aligned(16))) __bf16 vs[NP * HD];
  __shared__ float  Ss[NP * NP];
  __shared__ __bf16 Ps[NP * NP];
  const int tid = threadIdx.x;
  const size_t base = ((size_t)win * NHh + head) * NP * HD;
  // Async DMA of the three 4KB tiles into LDS (16B per lane per issue).
  for (int i = tid * 8; i < NP * HD; i += 128 * 8) {
    async_ld_b128(lds_off(&qs[i]), qg + base + i);
    async_ld_b128(lds_off(&ks[i]), kg + base + i);
    async_ld_b128(lds_off(&vs[i]), vg + base + i);
  }
  wait_async0();
  __syncthreads();
  const int wv = tid >> 5, lane = tid & 31, l15 = lane & 15;
  const int kh = (lane >> 4) << 3;
  const int mh = (lane >> 4) << 3;
  // ---- S = q k^T ----
  {
    const v16bf a = frag_ld(qs + (wv * 16 + l15) * HD + kh);
    const int wl = win & 63, wh = wl >> 3, wwi = wl & 7;
#pragma unroll
    for (int nt = 0; nt < 4; ++nt) {
      const v16bf bfr = frag_ld(ks + (nt * 16 + l15) * HD + kh);
      v8f accv = {};
      accv = wmma_bf16(a, bfr, accv);
#pragma unroll
      for (int r = 0; r < 8; ++r) {
        const int n = wv * 16 + mh + r;   // query
        const int m = nt * 16 + l15;      // key
        float o;
        if (m >= Nn) {
          o = -1e30f;                     // pad key columns -> softmax weight 0
        } else if (n >= Nn) {
          o = accv[r];                    // pad query rows, discarded downstream
        } else {
          const int r1 = n / 7, c1 = n % 7, r2 = m / 7, c2 = m % 7;
          const int idx = (r1 - r2 + 6) * 13 + (c1 - c2 + 6);
          const float bias = rel_tab[idx * NHh + head];
          const int h1 = wh * 7 + r1, w1 = wwi * 7 + c1;
          const int h2 = wh * 7 + r2, w2 = wwi * 7 + c2;
          const int reg1 = (h1 < 49 ? 0 : (h1 < 53 ? 1 : 2)) * 3 +
                           (w1 < 49 ? 0 : (w1 < 53 ? 1 : 2));
          const int reg2 = (h2 < 49 ? 0 : (h2 < 53 ? 1 : 2)) * 3 +
                           (w2 < 49 ? 0 : (w2 < 53 ? 1 : 2));
          o = accv[r] + bias + (reg1 != reg2 ? -100.0f : 0.0f);
        }
        Ss[n * NP + m] = o;
      }
    }
  }
  __syncthreads();
  // ---- softmax over keys (rows independent) ----
  if (tid < NP) {
    float mx = -3e30f;
    for (int m = 0; m < NP; ++m) mx = fmaxf(mx, Ss[tid * NP + m]);
    float sum = 0.f;
    for (int m = 0; m < NP; ++m) {
      const float e = __expf(Ss[tid * NP + m] - mx);
      Ss[tid * NP + m] = e;
      sum += e;
    }
    const float inv = 1.0f / sum;
    for (int m = 0; m < NP; ++m)
      Ps[tid * NP + m] = (__bf16)(Ss[tid * NP + m] * inv);
  }
  __syncthreads();
  // ---- O = P @ v ----
#pragma unroll
  for (int dt = 0; dt < 2; ++dt) {
    v8f o = {};
#pragma unroll
    for (int kstep = 0; kstep < 2; ++kstep) {
      const v16bf pa = frag_ld(Ps + (wv * 16 + l15) * NP + kstep * 32 + kh);
      v16bf vb;
#pragma unroll
      for (int e = 0; e < 16; ++e) {
        const int m = kstep * 32 + kh + (e & 7) + ((e >> 3) << 4);
        vb[e] = vs[m * HD + dt * 16 + l15];
      }
      o = wmma_bf16(pa, vb, o);
    }
#pragma unroll
    for (int r = 0; r < 8; ++r) {
      const int n = wv * 16 + mh + r;
      const int d = dt * 16 + l15;
      xw[((size_t)win * NP + n) * Cc + head * HD + d] = (__bf16)o[r];
    }
  }
}

// ---------------------------------------------------------------------------
// K4: proj GEMM + window-reverse + roll(+3,+3) + residual -> d_out (f32 x)
// ---------------------------------------------------------------------------
__global__ void __launch_bounds__(128)
proj_gemm_kernel(const __bf16* __restrict__ A, const __bf16* __restrict__ W,
                 const float* __restrict__ bias, const float* __restrict__ x1,
                 float* __restrict__ outp) {
  const int lane = threadIdx.x & 31, wave = threadIdx.x >> 5;
  const int m0 = blockIdx.x * 32;
  const int n0 = (blockIdx.y * 4 + wave) * 64;
  v8f acc[2][4] = {};
  gemm32x64(A, W, Cc, m0, n0, lane, acc);
  const int l15 = lane & 15, mh = (lane >> 4) << 3;
#pragma unroll
  for (int mi = 0; mi < 2; ++mi)
#pragma unroll
    for (int ni = 0; ni < 4; ++ni)
#pragma unroll
      for (int r = 0; r < 8; ++r) {
        const int gm = m0 + mi * 16 + mh + r;
        const int gn = n0 + ni * 16 + l15;
        const int win = gm >> 6, row = gm & 63;
        if (row < Nn) {
          const float val = acc[mi][ni][r] + bias[gn];
          const int bb = win >> 6, wl = win & 63;
          const int wh = wl >> 3, wwi = wl & 7;
          const int rr = row / 7, cc = row % 7;
          const int h = (wh * 7 + rr + SSs) % Hh;
          const int w = (wwi * 7 + cc + SSs) % Ww;
          const size_t t = ((size_t)bb * (Hh * Ww) + h * Ww + w) * Cc + gn;
          outp[t] = x1[t] + val;   // shortcut + attention branch
        }
      }
}

// ---------------------------------------------------------------------------
// K6: fc1 GEMM + exact GELU -> hbf bf16 [50176][2048]
// ---------------------------------------------------------------------------
__global__ void __launch_bounds__(128)
fc1_gemm_kernel(const __bf16* __restrict__ A, const __bf16* __restrict__ W,
                const float* __restrict__ bias, __bf16* __restrict__ hout) {
  const int lane = threadIdx.x & 31, wave = threadIdx.x >> 5;
  const int m0 = blockIdx.x * 32;
  const int n0 = (blockIdx.y * 4 + wave) * 64;
  v8f acc[2][4] = {};
  gemm32x64(A, W, Cc, m0, n0, lane, acc);
  const int l15 = lane & 15, mh = (lane >> 4) << 3;
#pragma unroll
  for (int mi = 0; mi < 2; ++mi)
#pragma unroll
    for (int ni = 0; ni < 4; ++ni)
#pragma unroll
      for (int r = 0; r < 8; ++r) {
        const int gm = m0 + mi * 16 + mh + r;
        const int gn = n0 + ni * 16 + l15;
        float val = acc[mi][ni][r] + bias[gn];
        val = 0.5f * val * (1.0f + erff(val * 0.70710678118654752f));
        hout[(size_t)gm * MLPH + gn] = (__bf16)val;
      }
}

// ---------------------------------------------------------------------------
// K7: fc2 GEMM + residual (d_out holds x; add y in place)
// ---------------------------------------------------------------------------
__global__ void __launch_bounds__(128)
fc2_gemm_kernel(const __bf16* __restrict__ A, const __bf16* __restrict__ W,
                const float* __restrict__ bias, float* __restrict__ outp) {
  const int lane = threadIdx.x & 31, wave = threadIdx.x >> 5;
  const int m0 = blockIdx.x * 32;
  const int n0 = (blockIdx.y * 4 + wave) * 64;
  v8f acc[2][4] = {};
  gemm32x64(A, W, MLPH, m0, n0, lane, acc);
  const int l15 = lane & 15, mh = (lane >> 4) << 3;
#pragma unroll
  for (int mi = 0; mi < 2; ++mi)
#pragma unroll
    for (int ni = 0; ni < 4; ++ni)
#pragma unroll
      for (int r = 0; r < 8; ++r) {
        const int gm = m0 + mi * 16 + mh + r;
        const int gn = n0 + ni * 16 + l15;
        const float val = acc[mi][ni][r] + bias[gn];
        outp[(size_t)gm * Cc + gn] += val;
      }
}

// ---------------------------------------------------------------------------
// Launch
// ---------------------------------------------------------------------------
extern "C" void kernel_launch(void* const* d_in, const int* in_sizes, int n_in,
                              void* d_out, int out_size, void* d_ws,
                              size_t ws_size, hipStream_t stream) {
  (void)in_sizes; (void)n_in; (void)out_size; (void)ws_size;
  const float* x1    = (const float*)d_in[0];
  // d_in[1] (x2) is dead: it only feeds attn2/k2/v2 which only affect
  // xw[:, :, C:], and the reference discards those channels.
  const float* n1g   = (const float*)d_in[2];
  const float* n1b   = (const float*)d_in[3];
  const float* qkvw  = (const float*)d_in[4];
  const float* qkvb  = (const float*)d_in[5];
  const float* rel   = (const float*)d_in[6];
  const float* projw = (const float*)d_in[7];
  const float* projb = (const float*)d_in[8];
  const float* n2g   = (const float*)d_in[9];
  const float* n2b   = (const float*)d_in[10];
  const float* fc1w  = (const float*)d_in[11];
  const float* fc1b  = (const float*)d_in[12];
  const float* fc2w  = (const float*)d_in[13];
  const float* fc2b  = (const float*)d_in[14];
  float* out = (float*)d_out;

  char* ws = (char*)d_ws;
  size_t off = 0;
  auto carve = [&](size_t bytes) -> void* {
    void* p = ws + off;
    off = (off + bytes + 255) & ~(size_t)255;
    return p;
  };
  __bf16* wqkv  = (__bf16*)carve((size_t)3 * Cc * Cc * 2);
  __bf16* wproj = (__bf16*)carve((size_t)Cc * Cc * 2);
  __bf16* wfc1  = (__bf16*)carve((size_t)MLPH * Cc * 2);
  __bf16* wfc2  = (__bf16*)carve((size_t)Cc * MLPH * 2);
  __bf16* a1w   = (__bf16*)carve((size_t)MPAD * Cc * 2);
  __bf16* q1    = (__bf16*)carve((size_t)MPAD * Cc * 2);
  __bf16* k1    = (__bf16*)carve((size_t)MPAD * Cc * 2);
  __bf16* v1    = (__bf16*)carve((size_t)MPAD * Cc * 2);
  __bf16* ln2b  = (__bf16*)carve((size_t)TOK * Cc * 2);
  __bf16* hbf   = (__bf16*)carve((size_t)TOK * MLPH * 2);
  __bf16* xw = a1w;  // a1w is dead after the QKV GEMM; reuse for attn output

  // K0: weights -> bf16
  f32_to_bf16_kernel<<<(3 * Cc * Cc + 255) / 256, 256, 0, stream>>>(qkvw, wqkv, 3 * Cc * Cc);
  f32_to_bf16_kernel<<<(Cc * Cc + 255) / 256, 256, 0, stream>>>(projw, wproj, Cc * Cc);
  f32_to_bf16_kernel<<<(MLPH * Cc + 255) / 256, 256, 0, stream>>>(fc1w, wfc1, MLPH * Cc);
  f32_to_bf16_kernel<<<(Cc * MLPH + 255) / 256, 256, 0, stream>>>(fc2w, wfc2, Cc * MLPH);

  // K1: LN + shift + partition (pads zeroed)
  ln_roll_part_kernel<<<MPAD, 256, 0, stream>>>(x1, n1g, n1b, a1w);
  // K2: QKV GEMM (WMMA bf16)
  qkv_gemm_kernel<<<dim3(MPAD / 32, (3 * Cc) / 256), 128, 0, stream>>>(
      a1w, wqkv, qkvb, q1, k1, v1);
  // K3: windowed attention (async-LDS staging + WMMA bf16 + LDS softmax)
  attn_kernel<<<dim3(TWIN, NHh), 128, 0, stream>>>(q1, k1, v1, rel, xw);
  // K4: proj GEMM + reverse + unshift + residual -> x (in d_out)
  proj_gemm_kernel<<<dim3(MPAD / 32, Cc / 256), 128, 0, stream>>>(
      xw, wproj, projb, x1, out);
  // K5: LN2
  ln_kernel<<<TOK, 256, 0, stream>>>(out, n2g, n2b, ln2b);
  // K6: fc1 + GELU
  fc1_gemm_kernel<<<dim3(TOK / 32, MLPH / 256), 128, 0, stream>>>(
      ln2b, wfc1, fc1b, hbf);
  // K7: fc2 + residual -> out = x + y
  fc2_gemm_kernel<<<dim3(TOK / 32, Cc / 256), 128, 0, stream>>>(
      hbf, wfc2, fc2b, out);
}
